// KmeansQuantizer_6760278524431
// MI455X (gfx1250) — matrix operand order repaired and checked by reading, not compiled, not run on hardware
//
#include <hip/hip_runtime.h>

// ---------------------------------------------------------------------------
// KMeans quantizer for MI455X (gfx1250, wave32, WMMA).
//   distances = ||x||^2 - 2*(x . c) + ||c||^2     [N,K]
//   labels    = argmin_k distances                 [N] int64
//   preds     = codebook[labels]                   [N,D]
// Cross term via v_wmma_f32_16x16x32_bf16; x^2 / c^2 exact fp32 (precomputed).
// Streamed outputs use non-temporal stores so x/codebook stay L2-resident.
// ---------------------------------------------------------------------------

typedef __attribute__((ext_vector_type(16))) __bf16 v16bf;
typedef __attribute__((ext_vector_type(8)))  float  v8f;
typedef __attribute__((ext_vector_type(4)))  float  f4;

#define DIM     128
#define TM      64      // rows (x points) per block
#define TN      128     // cols (centroids) per block
#define XSTR    136     // padded ushort stride for bf16 tiles (bank-conflict free)
#define DSTR    132     // padded float stride for fp32 distance tile

// LDS layout (bytes):
//   [0      , 17408) : x tile  bf16  64 x 136          } aliased by
//   [17408  , 52224) : c tile  bf16 128 x 136          } dist tile 64x132 fp32 (33792 B)
//   [52224  , 52480) : xsq   float[64]
//   [52480  , 52992) : csq   float[128]
//   [52992  , 55040) : red   u64[256]
#define SM_C    17408
#define SM_XSQ  52224
#define SM_CSQ  52480
#define SM_RED  52992
#define SM_TOT  55040

union ABFrag { v16bf v; uint4 q[2]; };

__device__ __forceinline__ unsigned short f2bf(float f) {
    unsigned u = __float_as_uint(f);
    unsigned r = u + 0x7FFFu + ((u >> 16) & 1u);   // round-to-nearest-even
    return (unsigned short)(r >> 16);
}

__device__ __forceinline__ unsigned long long umin64(unsigned long long a,
                                                     unsigned long long b) {
    return a < b ? a : b;
}

// ---------------------------------------------------------------------------
__global__ void kmq_init(unsigned long long* __restrict__ rowmin, int n) {
    int i = blockIdx.x * blockDim.x + threadIdx.x;
    if (i < n) rowmin[i] = ~0ull;
}

// ---------------------------------------------------------------------------
// One wave per row: exact fp32 squared norms for x (rows [0,N)) and
// codebook (rows [N, N+K)). Deterministic shuffle-tree reduction.
__global__ __launch_bounds__(256)
void kmq_norms(const float* __restrict__ x, const float* __restrict__ cbook,
               float* __restrict__ xsq_g, float* __restrict__ csq_g, int N) {
    int r    = blockIdx.x * 8 + (threadIdx.x >> 5);
    int lane = threadIdx.x & 31;
    const float* src = (r < N) ? (x + (size_t)r * DIM)
                               : (cbook + (size_t)(r - N) * DIM);
    f4 v = *(const f4*)(src + lane * 4);
    float s = v.x * v.x + v.y * v.y + v.z * v.z + v.w * v.w;
#pragma unroll
    for (int m = 16; m >= 1; m >>= 1) s += __shfl_xor(s, m, 32);
    if (lane == 0) {
        if (r < N) xsq_g[r] = s;
        else       csq_g[r - N] = s;
    }
}

// ---------------------------------------------------------------------------
__global__ __launch_bounds__(256)
void kmq_main(const float* __restrict__ x, const float* __restrict__ cbook,
              const float* __restrict__ xsq_g, const float* __restrict__ csq_g,
              float* __restrict__ dists, unsigned long long* __restrict__ rowmin,
              int K) {
    __shared__ __align__(16) unsigned char smem[SM_TOT];
    unsigned short*     lx    = (unsigned short*)smem;
    unsigned short*     lc    = (unsigned short*)(smem + SM_C);
    float*              ldist = (float*)smem;                 // alias after barrier
    float*              xsq   = (float*)(smem + SM_XSQ);
    float*              csq   = (float*)(smem + SM_CSQ);
    unsigned long long* red   = (unsigned long long*)(smem + SM_RED);

    const int t    = threadIdx.x;
    const int lane = t & 31;
    const int wave = t >> 5;
    const int row0 = blockIdx.y * TM;
    const int k0   = blockIdx.x * TN;

    // ---------- Phase 1: stage tiles (fp32 -> bf16) + copy norms ----------
#pragma unroll
    for (int i = 0; i < 8; ++i) {              // 64x128 x-tile, coalesced f4
        int l4 = i * 256 + t;
        int r  = l4 >> 5;
        int c  = (l4 & 31) << 2;
        f4 v = *(const f4*)(x + (size_t)(row0 + r) * DIM + c);
        unsigned lo = (unsigned)f2bf(v.x) | ((unsigned)f2bf(v.y) << 16);
        unsigned hi = (unsigned)f2bf(v.z) | ((unsigned)f2bf(v.w) << 16);
        unsigned* p = (unsigned*)(lx + r * XSTR + c);
        p[0] = lo; p[1] = hi;
    }
#pragma unroll
    for (int i = 0; i < 16; ++i) {             // 128x128 c-tile
        int l4 = i * 256 + t;
        int r  = l4 >> 5;
        int c  = (l4 & 31) << 2;
        f4 v = *(const f4*)(cbook + (size_t)(k0 + r) * DIM + c);
        unsigned lo = (unsigned)f2bf(v.x) | ((unsigned)f2bf(v.y) << 16);
        unsigned hi = (unsigned)f2bf(v.z) | ((unsigned)f2bf(v.w) << 16);
        unsigned* p = (unsigned*)(lc + r * XSTR + c);
        p[0] = lo; p[1] = hi;
    }
    if (t < TM)               xsq[t]      = xsq_g[row0 + t];
    else if (t < TM + TN)     csq[t - TM] = csq_g[k0 + (t - TM)];
    __syncthreads();

    // ---------- Phase 2: cross term, 16x64 per wave via WMMA bf16 --------
    const int m0 = (wave & 3) * 16;            // row group
    const int n0 = (wave >> 2) * 64;           // col group (4 x 16)
    v8f acc[4];
#pragma unroll
    for (int j = 0; j < 4; ++j) acc[j] = (v8f)0.0f;

    const int hiHalf = (lane & 16) ? 1 : 0;
#pragma unroll
    for (int ks = 0; ks < 4; ++ks) {           // D = 4 * 32
        const int d0 = ks * 32;
        // A fragment: lane<16 -> K {0..7,16..23}, lane>=16 -> +8
        ABFrag a;
        {
            const int ar  = m0 + (lane & 15);
            const int akb = d0 + (hiHalf ? 8 : 0);
            a.q[0] = *(const uint4*)(lx + ar * XSTR + akb);
            a.q[1] = *(const uint4*)(lx + ar * XSTR + akb + 16);
        }
#pragma unroll
        for (int j = 0; j < 4; ++j) {
            // B fragment: lane holds centroid column, K {0..7,16..23}, lane>=16 -> +8
            ABFrag b;
            const int bc  = n0 + j * 16 + (lane & 15);
            const int bkb = d0 + (hiHalf ? 8 : 0);
            const uint4* bp = (const uint4*)(lc + bc * XSTR + bkb);
            b.q[0] = bp[0];
            b.q[1] = bp[2];
            acc[j] = __builtin_amdgcn_wmma_f32_16x16x32_bf16(
                false, a.v, false, b.v, (short)0, acc[j], false, false);
        }
    }

    // ---------- Phase 3: spill tile to LDS (aliases bf16 tiles) ----------
    __syncthreads();                           // everyone done reading lx/lc
#pragma unroll
    for (int j = 0; j < 4; ++j) {
#pragma unroll
        for (int v = 0; v < 8; ++v) {
            int rr = m0 + v + (hiHalf ? 8 : 0);
            int cc = n0 + j * 16 + (lane & 15);
            ldist[rr * DSTR + cc] = acc[j][v];
        }
    }
    __syncthreads();

    // coalesced streaming (non-temporal) global write of full distances
#pragma unroll
    for (int i = 0; i < 32; ++i) {
        int lin = i * 256 + t;
        int r   = lin >> 7;
        int c   = lin & 127;
        float val = xsq[r] + csq[c] - 2.0f * ldist[r * DSTR + c];
        __builtin_nontemporal_store(val, dists + (size_t)(row0 + r) * K + k0 + c);
    }

    // per-row argmin over this 128-col block -> packed {orderedKey, k} u64
    const int arow = t >> 2;
    const int cb0  = (t & 3) << 5;
    const float xs = xsq[arow];
    float best  = 3.402823466e38f;
    int   bestk = k0 + cb0;
#pragma unroll 8
    for (int c = 0; c < 32; ++c) {
        int col = cb0 + c;
        float v = xs + csq[col] - 2.0f * ldist[arow * DSTR + col];
        if (v < best) { best = v; bestk = k0 + col; }
    }
    unsigned u   = __float_as_uint(best);
    unsigned key = (u & 0x80000000u) ? ~u : (u | 0x80000000u);
    red[t] = ((unsigned long long)key << 32) | (unsigned)bestk;
    __syncthreads();
    if ((t & 3) == 0) {
        unsigned long long m = red[t];
        m = umin64(m, red[t + 1]);
        m = umin64(m, red[t + 2]);
        m = umin64(m, red[t + 3]);
        atomicMin(rowmin + row0 + arow, m);    // global_atomic_min_u64
    }
}

// ---------------------------------------------------------------------------
// One wave per row: decode label, gather centroid into preds, write int64 label
__global__ __launch_bounds__(256)
void kmq_finalize(const float* __restrict__ cbook,
                  const unsigned long long* __restrict__ rowmin,
                  float* __restrict__ preds, long long* __restrict__ labels) {
    int n    = blockIdx.x * 8 + (threadIdx.x >> 5);
    int lane = threadIdx.x & 31;
    unsigned long long p = rowmin[n];
    int k = (int)(unsigned)(p & 0xFFFFFFFFull);
    f4 v = *(const f4*)(cbook + (size_t)k * DIM + lane * 4);   // L2-hot gather
    __builtin_nontemporal_store(v, (f4*)(preds + (size_t)n * DIM + lane * 4));
    if (lane == 0) __builtin_nontemporal_store((long long)k, labels + n);
}

// ---------------------------------------------------------------------------
extern "C" void kernel_launch(void* const* d_in, const int* in_sizes, int n_in,
                              void* d_out, int out_size, void* d_ws, size_t ws_size,
                              hipStream_t stream) {
    (void)n_in; (void)out_size; (void)ws_size;
    const float* x  = (const float*)d_in[0];
    const float* cb = (const float*)d_in[1];
    const int N = in_sizes[0] / DIM;           // 131072
    const int K = in_sizes[1] / DIM;           // 1024

    // d_out: [preds f32 N*D][labels i64 N][distances f32 N*K], native sizes
    float*     preds  = (float*)d_out;
    long long* labels = (long long*)((char*)d_out + (size_t)N * DIM * sizeof(float));
    float*     dists  = (float*)((char*)labels + (size_t)N * sizeof(long long));

    // workspace: [rowmin u64 N][xsq f32 N][csq f32 K]
    unsigned long long* rowmin = (unsigned long long*)d_ws;
    float* xsq_g = (float*)((char*)d_ws + (size_t)N * sizeof(unsigned long long));
    float* csq_g = xsq_g + N;

    kmq_init<<<(N + 255) / 256, 256, 0, stream>>>(rowmin, N);
    kmq_norms<<<(N + K) / 8, 256, 0, stream>>>(x, cb, xsq_g, csq_g, N);

    dim3 grid(K / TN, N / TM);                 // 8 x 2048
    kmq_main<<<grid, 256, 0, stream>>>(x, cb, xsq_g, csq_g, dists, rowmin, K);

    kmq_finalize<<<N / 8, 256, 0, stream>>>(cb, rowmin, preds, labels);
}
